// STGCN_60000693125479
// MI455X (gfx1250) — matrix operand order repaired and checked by reading, not compiled, not run on hardware
//
#include <hip/hip_runtime.h>

typedef __attribute__((ext_vector_type(16))) _Float16     v16h;
typedef __attribute__((ext_vector_type(8)))  float        v8f;
typedef __attribute__((ext_vector_type(4)))  unsigned int v4u;

// ---------------------------------------------------------------------------
// WMMA helper: D(f32 16x16) = A(f16 16x32) * B(f16 32x16) + C
// ---------------------------------------------------------------------------
__device__ __forceinline__ v8f wmma16(v16h a, v16h b, v8f c) {
  return __builtin_amdgcn_wmma_f32_16x16x32_f16(
      /*neg_a=*/false, a, /*neg_b=*/false, b,
      /*c_mod=*/(short)0, c, /*reuse_a=*/false, /*reuse_b=*/false);
}

// A-matrix 16x32 f16 layout (ISA 7.12.2): lane&15 = M, element e -> K:
//   lanes 0-15 : K runs [0..7],[16..23] ; lanes 16-31: +8
__device__ __forceinline__ int a_k(int lane, int e) {
  return ((e < 8) ? e : e + 8) + ((lane & 16) ? 8 : 0);
}

// Branchless, vectorized fragment loads: two 16B loads each.
__device__ __forceinline__ v16h load_a_frag(const _Float16* base, int lane) {
  const int off = (lane & 16) ? 8 : 0;
  union { v4u u[2]; v16h h; } r;
  r.u[0] = *(const v4u*)(base + off);
  r.u[1] = *(const v4u*)(base + off + 16);
  return r.h;
}
// B-matrix 32x16 f16 layout: lane&15 = N; K order matches A: [0..7],[16..23], lanes 16-31: +8.
// With B stored K-contiguous per column (transposed weights), two 8-half runs.
__device__ __forceinline__ v16h load_b_frag(const _Float16* base, int lane) {
  const int off = (lane & 16) ? 8 : 0;
  union { v4u u[2]; v16h h; } r;
  r.u[0] = *(const v4u*)(base + off);
  r.u[1] = *(const v4u*)(base + off + 16);
  return r.h;
}

// ---------------------------------------------------------------------------
// Prep kernels
// ---------------------------------------------------------------------------
__global__ void cast_f16_kernel(const float* __restrict__ in,
                                _Float16* __restrict__ out, int n) {
  int i = blockIdx.x * blockDim.x + threadIdx.x;
  if (i < n) out[i] = (_Float16)in[i];
}

__global__ void pad_a_kernel(const float* __restrict__ A, _Float16* __restrict__ Ap) {
  int i = blockIdx.x, j = threadIdx.x;  // 64 x 64
  Ap[i * 64 + j] = (i < 51 && j < 51) ? (_Float16)A[i * 51 + j] : (_Float16)0.f;
}

// conv weights (3, KT, 64) f32 -> transposed (3, 64, Kpad) f16, zero-padded K
__global__ void prep_wT_kernel(const float* __restrict__ w, _Float16* __restrict__ wT,
                               int KT, int Kpad) {
  int idx = blockIdx.x * blockDim.x + threadIdx.x;
  int total = 3 * 64 * Kpad;
  if (idx >= total) return;
  int k = idx % Kpad, n = (idx / Kpad) % 64, j = idx / (Kpad * 64);
  wT[idx] = (k < KT) ? (_Float16)w[(j * KT + k) * 64 + n] : (_Float16)0.f;
}

// theta (64,16) -> thT (16,64)
__global__ void prep_thT_kernel(const float* __restrict__ th, _Float16* __restrict__ thT) {
  int idx = blockIdx.x * blockDim.x + threadIdx.x;
  if (idx >= 16 * 64) return;
  int k = idx % 64, n = idx / 64;
  thT[idx] = (_Float16)th[k * 16 + n];
}

// fully_w (7040,25) -> fwT (32,7040), zero pad n>=25
__global__ void prep_fwT_kernel(const float* __restrict__ w, _Float16* __restrict__ wT) {
  int idx = blockIdx.x * blockDim.x + threadIdx.x;
  if (idx >= 32 * 7040) return;
  int k = idx % 7040, n = idx / 7040;
  wT[idx] = (n < 25) ? (_Float16)w[k * 25 + n] : (_Float16)0.f;
}

__global__ void zero_kernel(float* p, int n) {
  int i = blockIdx.x * blockDim.x + threadIdx.x;
  if (i < n) p[i] = 0.f;
}

// ---------------------------------------------------------------------------
// time_block: out = relu(conv0 + sigmoid(conv1) + conv2), k=3 temporal convs.
// GEMM: M = BN*Tout (multiple of 16), K = 3*CIN padded, 3 x 64 couts.
// 128 threads = 4 waves sharing one 16-row M tile; wave w owns couts [16w,16w+16).
// A tile staged via GLOBAL_LOAD_ASYNC_TO_LDS_B128 (ASYNCcnt); dual K-split
// accumulators per conv keep dependent WMMAs >=6 apart (no hazard NOPs).
// ---------------------------------------------------------------------------
template <int CIN>
__global__ void __launch_bounds__(128)
time_block_kernel(const _Float16* __restrict__ act,   // (BN, Tin, CIN) f16
                  const _Float16* __restrict__ wT,    // (3, 64, Kpad)  f16
                  const float* __restrict__ bias,     // (3, 64) f32
                  _Float16* __restrict__ out,         // (BN, Tout, 64) f16
                  int BN, int Tin) {
  constexpr int KT   = 3 * CIN;
  constexpr int Kpad = (KT + 31) & ~31;
  constexpr int SM   = 16 * (Kpad > 64 ? Kpad : 64);
  __shared__ _Float16 smem[SM];
  const int Tout = Tin - 2;

  const int tid  = threadIdx.x;
  const int wave = tid >> 5, lane = tid & 31;
  const int n    = wave * 16 + (lane & 15);
  const int mBase = blockIdx.x * 16;

  const _Float16* w0 = wT + (size_t)n * Kpad;
  const _Float16* w1 = w0 + (size_t)64 * Kpad;
  const _Float16* w2 = w1 + (size_t)64 * Kpad;

  // per-lane im2col row (used only by the CIN==3 scalar path)
  const int mRow = mBase + (lane & 15);
  const int bnL = mRow / Tout, tL = mRow % Tout;
  const _Float16* arowL = act + (size_t)bnL * Tin * CIN + (size_t)tL * CIN;

  if (CIN != 3) {
    // async cooperative A stage: 8 threads/row, 16B direct global->LDS copies
    const int row = tid >> 3, s0 = tid & 7;
    const int m = mBase + row;
    const int bn = m / Tout, t = m % Tout;
    const _Float16* arow = act + (size_t)bn * Tin * CIN + (size_t)t * CIN;
#pragma unroll
    for (int s = s0; s < Kpad / 8; s += 8) {
      unsigned ldsa = (unsigned)(size_t)&smem[row * Kpad + s * 8];
      asm volatile("global_load_async_to_lds_b128 %0, %1, off"
                   :: "v"(ldsa), "v"(arow + s * 8) : "memory");
    }
    asm volatile("s_wait_asynccnt 0x0" ::: "memory");
    __syncthreads();
  }
  const _Float16* As = &smem[(lane & 15) * Kpad];

  auto loadA = [&](int k0) {
    v16h f;
    if (CIN == 3) {  // misaligned rows: unconditional scalar gather (tiny stage)
#pragma unroll
      for (int e = 0; e < 16; ++e) f[e] = arowL[k0 + a_k(lane, e)];
    } else {
      f = load_a_frag(As + k0, lane);
    }
    return f;
  };

  v8f c0a = {}, c1a = {}, c2a = {}, c0b = {}, c1b = {}, c2b = {};
#pragma unroll
  for (int k0 = 0; k0 < Kpad; k0 += 64) {
    v16h afa = loadA(k0);
    c0a = wmma16(afa, load_b_frag(w0 + k0, lane), c0a);
    c1a = wmma16(afa, load_b_frag(w1 + k0, lane), c1a);
    c2a = wmma16(afa, load_b_frag(w2 + k0, lane), c2a);
    if (k0 + 32 < Kpad) {
      v16h afb = loadA(k0 + 32);
      c0b = wmma16(afb, load_b_frag(w0 + k0 + 32, lane), c0b);
      c1b = wmma16(afb, load_b_frag(w1 + k0 + 32, lane), c1b);
      c2b = wmma16(afb, load_b_frag(w2 + k0 + 32, lane), c2b);
    }
  }
  if (CIN != 3) __syncthreads();  // before smem reuse by epilogue

  v8f acc0 = c0a + c0b, acc1 = c1a + c1b, acc2 = c2a + c2b;

  // fused epilogue -> stage 16x64 tile in LDS -> coalesced 16B stores
  const float b0 = bias[n], b1 = bias[64 + n], b2 = bias[128 + n];
#pragma unroll
  for (int r = 0; r < 8; ++r) {
    float c1 = acc1[r] + b1;
    float sg = 1.f / (1.f + __expf(-c1));
    float v  = acc0[r] + b0 + sg + acc2[r] + b2;
    v = v > 0.f ? v : 0.f;
    smem[(r + ((lane & 16) ? 8 : 0)) * 64 + n] = (_Float16)v;
  }
  __syncthreads();
  {
    const int row = tid >> 3, s = tid & 7;
    *(v4u*)(out + (size_t)(mBase + row) * 64 + s * 8) = *(const v4u*)(&smem[row * 64 + s * 8]);
  }
}

// ---------------------------------------------------------------------------
// theta GEMM: Y = X @ theta, X: (M,64) f16, thT: (16,64) f16 -> (M,16) f16
// M multiple of 64: no tail guards. Independent K-split accumulators.
// ---------------------------------------------------------------------------
__global__ void __launch_bounds__(128)
theta_kernel(const _Float16* __restrict__ act, const _Float16* __restrict__ thT,
             _Float16* __restrict__ out) {
  __shared__ _Float16 ep[4 * 256];
  const int wave = threadIdx.x >> 5, lane = threadIdx.x & 31;
  const int mBase = (blockIdx.x * 4 + wave) * 16;
  const _Float16* arow = act + (size_t)(mBase + (lane & 15)) * 64;
  const int nn = lane & 15;
  const _Float16* brow = thT + (size_t)nn * 64;

  v8f ca = {}, cb = {};
  ca = wmma16(load_a_frag(arow, lane), load_b_frag(brow, lane), ca);
  cb = wmma16(load_a_frag(arow + 32, lane), load_b_frag(brow + 32, lane), cb);
  v8f acc = ca + cb;

  _Float16* w = &ep[wave * 256];
#pragma unroll
  for (int r = 0; r < 8; ++r)
    w[(r + ((lane & 16) ? 8 : 0)) * 16 + nn] = (_Float16)acc[r];
  __syncthreads();
  const int row = lane >> 1, s = lane & 1;
  *(v4u*)(out + (size_t)(mBase + row) * 16 + s * 8) = *(const v4u*)(&w[row * 16 + s * 8]);
}

// ---------------------------------------------------------------------------
// graph mix: out[b,i,t,:] = relu( sum_j A[i,j] * Y[b,j,t,:] ), 51 padded to 64.
// One block per (b,t). Y tile staged transposed in LDS; LDS-staged stores.
// ---------------------------------------------------------------------------
__global__ void __launch_bounds__(128)
graph_kernel(const _Float16* __restrict__ Ap,  // 64x64 zero-padded
             const _Float16* __restrict__ Y,   // (B,51,T,16)
             _Float16* __restrict__ out,       // (B,51,T,16)
             int T) {
  const int NN = 51;
  __shared__ _Float16 YsT[16 * 64];  // [c][j], zero-padded j
  __shared__ _Float16 ep[4 * 256];
  const int b = blockIdx.x / T, t = blockIdx.x % T;
  const int tid = threadIdx.x;

  {  // cooperative transposed stage: thread -> j = tid&63, c-group = tid>>6
    const int j = tid & 63, cg = tid >> 6;
    union { v4u u; _Float16 h[8]; } cv;
    if (j < NN) {
      cv.u = *(const v4u*)(Y + (((size_t)b * NN + j) * T + t) * 16 + cg * 8);
    } else {
#pragma unroll
      for (int c = 0; c < 8; ++c) cv.h[c] = (_Float16)0.f;
    }
#pragma unroll
    for (int c = 0; c < 8; ++c) YsT[(cg * 8 + c) * 64 + j] = cv.h[c];
  }
  __syncthreads();

  const int wave = tid >> 5, lane = tid & 31;
  const int iBase = wave * 16;
  const int iRow  = iBase + (lane & 15);  // < 64, Ap padded
  const int cc = lane & 15;

  v8f ca = {}, cb = {};
  ca = wmma16(load_a_frag(Ap + (size_t)iRow * 64, lane),
              load_b_frag(&YsT[cc * 64], lane), ca);
  cb = wmma16(load_a_frag(Ap + (size_t)iRow * 64 + 32, lane),
              load_b_frag(&YsT[cc * 64 + 32], lane), cb);
  v8f acc = ca + cb;

  _Float16* w = &ep[wave * 256];
#pragma unroll
  for (int r = 0; r < 8; ++r) {
    float v = fmaxf(acc[r], 0.f);
    w[(r + ((lane & 16) ? 8 : 0)) * 16 + cc] = (_Float16)v;
  }
  __syncthreads();
  const int row = lane >> 1, s = lane & 1;
  const int i = iBase + row;
  if (i < NN)
    *(v4u*)(out + (((size_t)b * NN + i) * T + t) * 16 + s * 8) =
        *(const v4u*)(&w[row * 16 + s * 8]);
}

// ---------------------------------------------------------------------------
// BatchNorm over node dim: reduce sum/sumsq per node over (B,T,64)
// ---------------------------------------------------------------------------
__global__ void bn_reduce_kernel(const _Float16* __restrict__ x, float* __restrict__ sums,
                                 int T) {
  const int n = blockIdx.y, b = blockIdx.x;
  const v4u* p8 = (const v4u*)(x + (((size_t)b * 51 + n) * T) * 64);
  const int cnt8 = T * 8;
  float s1 = 0.f, s2 = 0.f;
  for (int i = threadIdx.x; i < cnt8; i += blockDim.x) {
    union { v4u u; _Float16 h[8]; } cv;
    cv.u = p8[i];
#pragma unroll
    for (int c = 0; c < 8; ++c) {
      float v = (float)cv.h[c];
      s1 += v; s2 += v * v;
    }
  }
  __shared__ float r1[256], r2[256];
  r1[threadIdx.x] = s1; r2[threadIdx.x] = s2;
  __syncthreads();
  for (int s = 128; s > 0; s >>= 1) {
    if ((int)threadIdx.x < s) { r1[threadIdx.x] += r1[threadIdx.x + s]; r2[threadIdx.x] += r2[threadIdx.x + s]; }
    __syncthreads();
  }
  if (threadIdx.x == 0) {
    atomicAdd(&sums[n * 2 + 0], r1[0]);
    atomicAdd(&sums[n * 2 + 1], r2[0]);
  }
}

__global__ void bn_apply_kernel(const _Float16* __restrict__ x, const float* __restrict__ sums,
                                const float* __restrict__ gamma, const float* __restrict__ beta,
                                _Float16* __restrict__ out, int Bsz, int T) {
  size_t total8 = (size_t)Bsz * 51 * T * 8;  // vectors of 8 halves
  size_t i = (size_t)blockIdx.x * blockDim.x + threadIdx.x;
  if (i >= total8) return;
  size_t per8 = (size_t)T * 8;
  int n = (int)((i / per8) % 51);
  float cnt = (float)Bsz * (float)T * 64.f;
  float mu  = sums[n * 2 + 0] / cnt;
  float var = sums[n * 2 + 1] / cnt - mu * mu;
  float sc  = rsqrtf(var + 1e-5f) * gamma[n];
  float bt  = beta[n];
  union { v4u u; _Float16 h[8]; } cv;
  cv.u = ((const v4u*)x)[i];
#pragma unroll
  for (int c = 0; c < 8; ++c) cv.h[c] = (_Float16)(((float)cv.h[c] - mu) * sc + bt);
  ((v4u*)out)[i] = cv.u;
}

// ---------------------------------------------------------------------------
// fully: (M,7040) @ fwT(32,7040) + bias -> (M,25) f16  (M multiple of 64)
// 4 independent accumulators (2 N-tiles x 2 K-split).
// ---------------------------------------------------------------------------
__global__ void __launch_bounds__(128)
fully_kernel(const _Float16* __restrict__ act, const _Float16* __restrict__ wT,
             const float* __restrict__ bias, _Float16* __restrict__ out, int K) {
  const int wave = threadIdx.x >> 5, lane = threadIdx.x & 31;
  const int mBase = (blockIdx.x * 4 + wave) * 16;
  const _Float16* arow = act + (size_t)(mBase + (lane & 15)) * K;
  const int nn = lane & 15;
  const _Float16* b0row = wT + (size_t)nn * K;
  const _Float16* b1row = wT + (size_t)(16 + nn) * K;

  v8f c0a = {}, c1a = {}, c0b = {}, c1b = {};
  for (int k0 = 0; k0 < K; k0 += 64) {
    __builtin_prefetch(arow + k0 + 128, 0, 0);
    v16h a0 = load_a_frag(arow + k0, lane);
    v16h a1 = load_a_frag(arow + k0 + 32, lane);
    c0a = wmma16(a0, load_b_frag(b0row + k0, lane), c0a);
    c1a = wmma16(a0, load_b_frag(b1row + k0, lane), c1a);
    c0b = wmma16(a1, load_b_frag(b0row + k0 + 32, lane), c0b);
    c1b = wmma16(a1, load_b_frag(b1row + k0 + 32, lane), c1b);
  }
  v8f acc0 = c0a + c0b, acc1 = c1a + c1b;
#pragma unroll
  for (int r = 0; r < 8; ++r) {
    int m = mBase + r + ((lane & 16) ? 8 : 0);
    if (nn < 25)      out[(size_t)m * 25 + nn]      = (_Float16)(acc0[r] + bias[nn]);
    if (16 + nn < 25) out[(size_t)m * 25 + 16 + nn] = (_Float16)(acc1[r] + bias[16 + nn]);
  }
}

// ---------------------------------------------------------------------------
// fc: (B, 1275) @ (1275, 8) + b -> (B, 8) f32   (tiny, scalar)
// ---------------------------------------------------------------------------
__global__ void fc_kernel(const _Float16* __restrict__ o4, const float* __restrict__ w,
                          const float* __restrict__ bias, float* __restrict__ out) {
  const int b = blockIdx.x;
  const int cls = threadIdx.x & 7, grp = threadIdx.x >> 3;  // 64 threads: 8x8
  float s = 0.f;
  for (int k = grp; k < 1275; k += 8)
    s += (float)o4[(size_t)b * 1275 + k] * w[k * 8 + cls];
  __shared__ float red[64];
  red[threadIdx.x] = s;
  __syncthreads();
  if (grp == 0) {
    float tot = bias[cls];
#pragma unroll
    for (int g = 0; g < 8; ++g) tot += red[g * 8 + cls];
    out[b * 8 + cls] = tot;
  }
}

// ---------------------------------------------------------------------------
// Host-side orchestration
// ---------------------------------------------------------------------------
extern "C" void kernel_launch(void* const* d_in, const int* in_sizes, int n_in,
                              void* d_out, int out_size, void* d_ws, size_t ws_size,
                              hipStream_t stream) {
  (void)in_sizes; (void)n_in; (void)out_size; (void)ws_size;
  const int B = 256, N = 51, T0 = 120, F = 3;
  const int BN = B * N;  // 13056, multiple of 64 -> exact tiling everywhere

  const float* A_hat   = (const float*)d_in[0];
  const float* X       = (const float*)d_in[1];
  const float* b1_tw   = (const float*)d_in[2];
  const float* b1_tb   = (const float*)d_in[3];
  const float* b1_th   = (const float*)d_in[4];
  const float* b1_t2w  = (const float*)d_in[5];
  const float* b1_t2b  = (const float*)d_in[6];
  const float* b1_bng  = (const float*)d_in[7];
  const float* b1_bnb  = (const float*)d_in[8];
  const float* b2_tw   = (const float*)d_in[9];
  const float* b2_tb   = (const float*)d_in[10];
  const float* b2_th   = (const float*)d_in[11];
  const float* b2_t2w  = (const float*)d_in[12];
  const float* b2_t2b  = (const float*)d_in[13];
  const float* b2_bng  = (const float*)d_in[14];
  const float* b2_bnb  = (const float*)d_in[15];
  const float* lt_w    = (const float*)d_in[16];
  const float* lt_b    = (const float*)d_in[17];
  const float* fully_w = (const float*)d_in[18];
  const float* fully_b = (const float*)d_in[19];
  const float* fc_w    = (const float*)d_in[20];
  const float* fc_b    = (const float*)d_in[21];
  float* out = (float*)d_out;

  // workspace carve-up
  char* ws = (char*)d_ws;
  size_t off = 0;
  auto alloc = [&](size_t bytes) -> void* {
    void* p = ws + off;
    off = (off + bytes + 255) & ~(size_t)255;
    return p;
  };
  _Float16* x16   = (_Float16*)alloc((size_t)BN * T0 * F * 2);
  _Float16* bufA  = (_Float16*)alloc((size_t)BN * T0 * 64 * 2);
  _Float16* bufB  = (_Float16*)alloc((size_t)BN * T0 * 64 * 2);
  _Float16* bufC  = (_Float16*)alloc((size_t)BN * T0 * 16 * 2);
  _Float16* bufD  = (_Float16*)alloc((size_t)BN * T0 * 16 * 2);
  _Float16* w1aT  = (_Float16*)alloc((size_t)3 * 64 * 32 * 2);   // Kpad(9)=32
  _Float16* w1bT  = (_Float16*)alloc((size_t)3 * 64 * 64 * 2);   // Kpad(48)=64
  _Float16* w2aT  = (_Float16*)alloc((size_t)3 * 64 * 192 * 2);  // Kpad(192)=192
  _Float16* w2bT  = (_Float16*)alloc((size_t)3 * 64 * 64 * 2);
  _Float16* wltT  = (_Float16*)alloc((size_t)3 * 64 * 192 * 2);
  _Float16* th1T  = (_Float16*)alloc(16 * 64 * 2);
  _Float16* th2T  = (_Float16*)alloc(16 * 64 * 2);
  _Float16* Apad  = (_Float16*)alloc(64 * 64 * 2);
  _Float16* fwT   = (_Float16*)alloc((size_t)32 * 7040 * 2);
  _Float16* o4buf = (_Float16*)alloc((size_t)BN * 25 * 2);
  float*    sums  = (float*)alloc(51 * 2 * sizeof(float));
  (void)alloc(4096);  // tail pad: padded-K overreads stay in-bounds

  // ---- prep ----
  {
    int n = BN * T0 * F;
    cast_f16_kernel<<<(n + 255) / 256, 256, 0, stream>>>(X, x16, n);
  }
  prep_wT_kernel<<<(3 * 64 * 32 + 255) / 256, 256, 0, stream>>>(b1_tw, w1aT, 9, 32);
  prep_wT_kernel<<<(3 * 64 * 64 + 255) / 256, 256, 0, stream>>>(b1_t2w, w1bT, 48, 64);
  prep_wT_kernel<<<(3 * 64 * 192 + 255) / 256, 256, 0, stream>>>(b2_tw, w2aT, 192, 192);
  prep_wT_kernel<<<(3 * 64 * 64 + 255) / 256, 256, 0, stream>>>(b2_t2w, w2bT, 48, 64);
  prep_wT_kernel<<<(3 * 64 * 192 + 255) / 256, 256, 0, stream>>>(lt_w, wltT, 192, 192);
  prep_thT_kernel<<<4, 256, 0, stream>>>(b1_th, th1T);
  prep_thT_kernel<<<4, 256, 0, stream>>>(b2_th, th2T);
  pad_a_kernel<<<64, 64, 0, stream>>>(A_hat, Apad);
  prep_fwT_kernel<<<(32 * 7040 + 255) / 256, 256, 0, stream>>>(fully_w, fwT);

  auto run_bn = [&](const _Float16* in, const float* g, const float* bt,
                    _Float16* o, int T) {
    zero_kernel<<<1, 128, 0, stream>>>(sums, 102);
    bn_reduce_kernel<<<dim3(B, 51), 256, 0, stream>>>(in, sums, T);
    size_t total8 = (size_t)B * 51 * T * 8;
    bn_apply_kernel<<<(unsigned)((total8 + 255) / 256), 256, 0, stream>>>(in, sums, g, bt, o, B, T);
  };

  // ---- block 1 ----  T: 120 -> 118 -> 116
  time_block_kernel<3><<<BN * 118 / 16, 128, 0, stream>>>(x16, w1aT, b1_tb, bufA, BN, 120);
  theta_kernel<<<BN * 118 / 64, 128, 0, stream>>>(bufA, th1T, bufC);
  graph_kernel<<<B * 118, 128, 0, stream>>>(Apad, bufC, bufD, 118);
  time_block_kernel<16><<<BN * 116 / 16, 128, 0, stream>>>(bufD, w1bT, b1_t2b, bufB, BN, 118);
  run_bn(bufB, b1_bng, b1_bnb, bufA, 116);

  // ---- block 2 ----  T: 116 -> 114 -> 112
  time_block_kernel<64><<<BN * 114 / 16, 128, 0, stream>>>(bufA, w2aT, b2_tb, bufB, BN, 116);
  theta_kernel<<<BN * 114 / 64, 128, 0, stream>>>(bufB, th2T, bufC);
  graph_kernel<<<B * 114, 128, 0, stream>>>(Apad, bufC, bufD, 114);
  time_block_kernel<16><<<BN * 112 / 16, 128, 0, stream>>>(bufD, w2bT, b2_t2b, bufA, BN, 114);
  run_bn(bufA, b2_bng, b2_bnb, bufB, 112);

  // ---- last temporal ----  T: 112 -> 110
  time_block_kernel<64><<<BN * 110 / 16, 128, 0, stream>>>(bufB, wltT, lt_b, bufA, BN, 112);

  // ---- fully: (BN, 110*64) @ (7040, 25) ----
  fully_kernel<<<BN / 64, 128, 0, stream>>>(bufA, fwT, fully_b, o4buf, 7040);

  // ---- fc: (B, 1275) @ (1275, 8) ----
  fc_kernel<<<B, 64, 0, stream>>>(o4buf, fc_w, fc_b, out);
}
